// LSTMModel_12764642803974
// MI455X (gfx1250) — matrix operand order repaired and checked by reading, not compiled, not run on hardware
//
#include <hip/hip_runtime.h>
#include <cstdint>
#include <cstddef>

// ---------------------------------------------------------------------------
// MI455X (gfx1250) 2-layer LSTM + MLP head.
// v3: weights register-resident (K split across wave pairs, LDS reduction),
//     TDM double-buffered streaming of x into LDS, fast exp2/rcp activations.
// ---------------------------------------------------------------------------

typedef __attribute__((ext_vector_type(16))) __bf16 v16bf;
typedef __attribute__((ext_vector_type(8)))  float  v8f;
typedef __attribute__((ext_vector_type(4)))  unsigned int v4u;
typedef __attribute__((ext_vector_type(8)))  int    v8i_t;
typedef __attribute__((ext_vector_type(4)))  int    v4i_t;

#if defined(__clang__) && __has_builtin(__builtin_amdgcn_tensor_load_to_lds)
#define USE_TDM 1
#else
#define USE_TDM 0
#endif

static __device__ __forceinline__ unsigned short f2bf(float f) {
  unsigned u = __float_as_uint(f);
  u += 0x7FFFu + ((u >> 16) & 1u);            // round-to-nearest-even
  return (unsigned short)(u >> 16);
}
static __device__ __forceinline__ float bf2f(unsigned short h) {
  return __uint_as_float(((unsigned)h) << 16);
}

// sigmoid/tanh on the raw transcendental pipe: v_exp_f32 + v_rcp_f32
static __device__ __forceinline__ float fsigmoid(float x) {
  float e = __builtin_amdgcn_exp2f(-1.4426950408889634f * x);
  return __builtin_amdgcn_rcpf(1.f + e);
}
static __device__ __forceinline__ float ftanh(float x) {
  float e = __builtin_amdgcn_exp2f(2.8853900817779268f * x);
  return 1.f - 2.f * __builtin_amdgcn_rcpf(e + 1.f);
}

union Frag16 { v16bf v; uint4 q[2]; };

// ---------------------------------------------------------------------------
__global__ void k_f32_to_bf16(const float* __restrict__ src,
                              unsigned short* __restrict__ dst, int n) {
  int i = blockIdx.x * blockDim.x + threadIdx.x;
  if (i < n) dst[i] = f2bf(src[i]);
}

__global__ void k_bias(const float* __restrict__ a, const float* __restrict__ b,
                       float* __restrict__ o, int n) {
  int i = blockIdx.x * blockDim.x + threadIdx.x;
  if (i < n) o[i] = a[i] + b[i];
}

// ---------------------------------------------------------------------------
// Pre-pack [W_ih | W_hh] into per-lane WMMA B-fragment order:
// frag(kt,nt) = 32 lanes x 16 bf16 contiguous; lane l holds column
// n = nt*16 + (l&15); element i holds K = kt*32 + (l>>4)*16 + i.
// ---------------------------------------------------------------------------
template <int DIN>
__global__ void k_prepack(const float* __restrict__ Wih,
                          const float* __restrict__ Whh,
                          unsigned short* __restrict__ out) {
  constexpr int KTOT = DIN + 128;
  constexpr int KT   = KTOT / 32;
  int idx = blockIdx.x * blockDim.x + threadIdx.x;
  if (idx >= KT * 32 * 32 * 16) return;
  int i  = idx & 15;
  int ln = (idx >> 4) & 31;
  int nt = (idx >> 9) & 31;
  int kt = idx >> 14;
  int n = nt * 16 + (ln & 15);
  int k = kt * 32 + (ln >> 4) * 16 + i;
  float v = (k < DIN) ? Wih[n * DIN + k] : Whh[n * 128 + (k - DIN)];
  out[idx] = f2bf(v);
}

// ---------------------------------------------------------------------------
// Persistent LSTM layer. Grid = B/16 WGs, 512 threads (16 waves).
// Wave pair (w, w+8) owns hidden slice j in [w*16, w*16+16) across 4 gates;
// low wave does k-tiles [0, KT/2), high wave [KT/2, KT). Weights live in
// VGPRs. Partial sums merged via LDS. x_t streamed by TDM, double-buffered.
// ---------------------------------------------------------------------------
template <int DIN, bool STORE_SEQ>
__global__ void __launch_bounds__(512)
k_lstm(const unsigned short* __restrict__ Xseq,   // (B, 512, DIN) bf16
       const unsigned short* __restrict__ Wpk,    // packed B fragments
       const float* __restrict__ bias,            // 512 fused biases
       unsigned short* __restrict__ Yout)         // seq (B,512,128) or last (B,128)
{
  constexpr int T   = 512;
  constexpr int KT  = (DIN + 128) / 32;
  constexpr int KTH = KT / 2;
  static_assert(KT % 2 == 0, "K tiles must split across wave pairs");

  __shared__ unsigned short Xs[2][16][DIN];   // double-buffered x tile
  __shared__ unsigned short Hs[16][128];      // recurrent h tile (bf16)
  __shared__ float Red[8][4][16][16];         // [pair][gate][ncol][m] partials

  const int tid   = threadIdx.x;
  const int wv    = tid >> 5;
  const int ln    = tid & 31;
  const int w8    = wv & 7;              // pair id / hidden slice
  const int khalf = wv >> 3;             // 0 = low half of K, 1 = high half
  const int b0    = blockIdx.x * 16;
  const int j     = w8 * 16 + (ln & 15); // hidden column owned by this lane
  const int mh    = (ln >> 4) << 3;      // C/D tile row offset (0 or 8)
  const int mrow  = ln & 15;             // A-fragment row
  const int half8 = (ln >> 4) * 16;      // A-fragment K sub-offset
  const int nc    = ln & 15;             // C/D tile column (within slice)

  // ---- load this wave's B fragments from global into registers (once) ----
  Frag16 breg[KTH][4];
#pragma unroll
  for (int kk = 0; kk < KTH; ++kk)
#pragma unroll
    for (int g = 0; g < 4; ++g) {
      const char* bb = (const char*)Wpk +
          ((((khalf * KTH + kk) * 32) + (g * 8 + w8)) * 32 + ln) * 32;
      breg[kk][g].q[0] = *(const uint4*)bb;
      breg[kk][g].q[1] = *(const uint4*)(bb + 16);
    }

  // zero h buffer
  for (int o = tid; o < 16 * 64; o += 512)
    ((unsigned int*)&Hs[0][0])[o] = 0u;

#if USE_TDM
  auto tdm_issue = [&](int t, int buf) {
    uint64_t ga = (uint64_t)(uintptr_t)(Xseq + ((size_t)b0 * T + t) * DIN);
    unsigned lds = (unsigned)(uintptr_t)&Xs[buf][0][0];
    v4u g0;
    g0[0] = 1u;                                             // count=1
    g0[1] = lds;                                            // LDS dest
    g0[2] = (unsigned)(ga & 0xFFFFFFFFu);
    g0[3] = (unsigned)((ga >> 32) & 0x01FFFFFFu) | (2u << 30); // addr hi | type=2
    v8i_t g1;
    g1[0] = (int)(1u << 16);                 // data_size = 2B
    g1[1] = (int)((unsigned)DIN << 16);      // tensor_dim0 lo16 = DIN
    g1[2] = (int)(16u << 16);                // dim0 hi16 = 0 | tensor_dim1 = 16
    g1[3] = (int)((unsigned)DIN << 16);      // tile_dim0 = DIN
    g1[4] = 16;                              // tile_dim1 = 16, tile_dim2 = 0
    g1[5] = (int)(T * DIN);                  // tensor_dim0_stride (row pitch)
    g1[6] = 0; g1[7] = 0;
    v4i_t g2; g2[0] = 0; g2[1] = 0; g2[2] = 0; g2[3] = 0;
    v4i_t g3; g3[0] = 0; g3[1] = 0; g3[2] = 0; g3[3] = 0;
    v8i_t g4; g4[0] = 0; g4[1] = 0; g4[2] = 0; g4[3] = 0;
    g4[4] = 0; g4[5] = 0; g4[6] = 0; g4[7] = 0;
    __builtin_amdgcn_tensor_load_to_lds(g0, g1, g2, g3, g4, 0);
  };
  if (wv == 0) { tdm_issue(0, 0); tdm_issue(1, 1); }
#endif

  // per-lane recurrent state (low waves only)
  float cst[8];
  float bg[4];
  if (khalf == 0) {
#pragma unroll
    for (int v = 0; v < 8; ++v) cst[v] = 0.f;
#pragma unroll
    for (int g = 0; g < 4; ++g) bg[g] = bias[g * 128 + j];
  }

  for (int t = 0; t < T; ++t) {
    const int buf = t & 1;
#if USE_TDM
    if (wv == 0) __builtin_amdgcn_s_wait_tensorcnt(1);   // x[t] landed
#else
    for (int d = tid; d < 16 * (DIN / 2); d += 512) {
      int m = d / (DIN / 2), c = d % (DIN / 2);
      ((unsigned int*)&Xs[buf][m][0])[c] =
          ((const unsigned int*)(Xseq + ((size_t)(b0 + m) * T + t) * DIN))[c];
    }
#endif
    __syncthreads();                                     // (1) x + h visible

    // ---- this wave's half of the gate GEMM, B from registers ----
    v8f acc[4];
#pragma unroll
    for (int g = 0; g < 4; ++g)
#pragma unroll
      for (int v = 0; v < 8; ++v) acc[g][v] = 0.f;

#pragma unroll
    for (int kk = 0; kk < KTH; ++kk) {
      const int kcol = (khalf * KTH + kk) * 32 + half8;
      const unsigned short* ab = (kcol < DIN)
          ? (&Xs[buf][0][0] + mrow * DIN + kcol)
          : (&Hs[0][0] + mrow * 128 + (kcol - DIN));
      Frag16 af;
      af.q[0] = *(const uint4*)ab;
      af.q[1] = *(const uint4*)((const char*)ab + 16);
#pragma unroll
      for (int g = 0; g < 4; ++g)
        acc[g] = __builtin_amdgcn_wmma_f32_16x16x32_bf16(
            false, af.v, false, breg[kk][g].v, (short)0, acc[g], false, false);
    }
    __syncthreads();                                     // (2) LDS reads done

    if (khalf == 1) {                                    // publish partials
#pragma unroll
      for (int g = 0; g < 4; ++g) {
        float4* rp = (float4*)&Red[w8][g][nc][mh];
        rp[0] = make_float4(acc[g][0], acc[g][1], acc[g][2], acc[g][3]);
        rp[1] = make_float4(acc[g][4], acc[g][5], acc[g][6], acc[g][7]);
      }
    }
#if USE_TDM
    if (wv == 0) {                                       // prefetch x[t+2]
      int tc = (t + 2 < T) ? t + 2 : T - 1;
      tdm_issue(tc, buf);
    }
#endif
    __syncthreads();                                     // (3) partials visible

    if (khalf == 0) {
#pragma unroll
      for (int g = 0; g < 4; ++g) {
        const float4* rp = (const float4*)&Red[w8][g][nc][mh];
        float4 r0 = rp[0], r1 = rp[1];
        acc[g][0] += r0.x; acc[g][1] += r0.y; acc[g][2] += r0.z; acc[g][3] += r0.w;
        acc[g][4] += r1.x; acc[g][5] += r1.y; acc[g][6] += r1.z; acc[g][7] += r1.w;
      }
#pragma unroll
      for (int v = 0; v < 8; ++v) {
        float si = fsigmoid(acc[0][v] + bg[0]);
        float sf = fsigmoid(acc[1][v] + bg[1]);
        float tg = ftanh(acc[2][v] + bg[2]);
        float so = fsigmoid(acc[3][v] + bg[3]);
        float c  = sf * cst[v] + si * tg;
        cst[v]   = c;
        float h  = so * ftanh(c);
        unsigned short hb = f2bf(h);
        int m = mh + v;
        Hs[m][j] = hb;                                   // feed next timestep
        if (STORE_SEQ) {
          Yout[((size_t)(b0 + m) * T + t) * 128 + j] = hb;
        } else if (t == T - 1) {
          Yout[(size_t)(b0 + m) * 128 + j] = hb;
        }
      }
    }
  }
}

// ---------------------------------------------------------------------------
// MLP head: out = relu(last @ W1^T + b1) @ W2^T + b2  (256 x 3, fp32)
// ---------------------------------------------------------------------------
__global__ void k_head(const unsigned short* __restrict__ h1,
                       const float* __restrict__ W1, const float* __restrict__ b1,
                       const float* __restrict__ W2, const float* __restrict__ b2,
                       float* __restrict__ out) {
  int b = blockIdx.x * blockDim.x + threadIdx.x;
  if (b >= 256) return;
  float last[128];
#pragma unroll
  for (int k = 0; k < 128; ++k) last[k] = bf2f(h1[b * 128 + k]);
  float o0 = b2[0], o1 = b2[1], o2 = b2[2];
  for (int jj = 0; jj < 64; ++jj) {
    float s = b1[jj];
#pragma unroll 8
    for (int k = 0; k < 128; ++k) s += last[k] * W1[jj * 128 + k];
    s = fmaxf(s, 0.f);
    o0 += s * W2[jj];
    o1 += s * W2[64 + jj];
    o2 += s * W2[128 + jj];
  }
  out[b * 3 + 0] = o0;
  out[b * 3 + 1] = o1;
  out[b * 3 + 2] = o2;
}

// ---------------------------------------------------------------------------
extern "C" void kernel_launch(void* const* d_in, const int* in_sizes, int n_in,
                              void* d_out, int out_size, void* d_ws, size_t ws_size,
                              hipStream_t stream) {
  const float* x    = (const float*)d_in[0];
  const float* Wih0 = (const float*)d_in[1];
  const float* Whh0 = (const float*)d_in[2];
  const float* bih0 = (const float*)d_in[3];
  const float* bhh0 = (const float*)d_in[4];
  const float* Wih1 = (const float*)d_in[5];
  const float* Whh1 = (const float*)d_in[6];
  const float* bih1 = (const float*)d_in[7];
  const float* bhh1 = (const float*)d_in[8];
  const float* W1   = (const float*)d_in[9];
  const float* b1   = (const float*)d_in[10];
  const float* W2   = (const float*)d_in[11];
  const float* b2   = (const float*)d_in[12];
  float* out = (float*)d_out;

  char* ws = (char*)d_ws;
  const size_t BASE = (size_t)48 << 20;
  unsigned short* xbf    = (unsigned short*)(ws);                       // 16 MB
  unsigned short* h0seq  = (unsigned short*)(ws + ((size_t)16 << 20));  // 32 MB
  unsigned short* wpk0   = (unsigned short*)(ws + BASE);                // 192 KB
  unsigned short* wpk1   = (unsigned short*)(ws + BASE + 0x40000);      // 256 KB
  float*          bias0  = (float*)(ws + BASE + 0x80000);
  float*          bias1  = (float*)(ws + BASE + 0x80800);
  unsigned short* h1last = (unsigned short*)(ws + BASE + 0x81000);      // 64 KB

  const int nx = 256 * 512 * 64;
  k_f32_to_bf16<<<(nx + 255) / 256, 256, 0, stream>>>(x, xbf, nx);
  k_prepack<64><<<(6 * 32 * 32 * 16 + 255) / 256, 256, 0, stream>>>(Wih0, Whh0, wpk0);
  k_prepack<128><<<(8 * 32 * 32 * 16 + 255) / 256, 256, 0, stream>>>(Wih1, Whh1, wpk1);
  k_bias<<<2, 256, 0, stream>>>(bih0, bhh0, bias0, 512);
  k_bias<<<2, 256, 0, stream>>>(bih1, bhh1, bias1, 512);

  k_lstm<64, true><<<16, 512, 0, stream>>>(xbf, wpk0, bias0, h0seq);
  k_lstm<128, false><<<16, 512, 0, stream>>>(h0seq, wpk1, bias1, h1last);
  k_head<<<4, 64, 0, stream>>>(h1last, W1, b1, W2, b2, out);
}
